// GatedGINModel_64682207478382
// MI455X (gfx1250) — matrix operand order repaired and checked by reading, not compiled, not run on hardware
//
#include <hip/hip_runtime.h>

typedef __attribute__((ext_vector_type(16))) __bf16 v16bf;
typedef __attribute__((ext_vector_type(8)))  float  v8f;

#define HD      128
#define BN_EPS  1e-5f

// ---- dynamic LDS layout (bytes); every region offset is 32B-aligned ----
#define WROWB     288                        // 144 bf16 per weight row (bank-staggered)
#define OFF_WHI   0
#define OFF_WLO   (128 * WROWB)              // 36864
#define OFF_A     (2 * 128 * WROWB)          // 73728
#define AROWB     528                        // 132 floats per A row (bank-staggered)
#define OFF_SCALE (OFF_A + 128 * AROWB)      // 141312
#define OFF_SHIFT (OFF_SCALE + 512)
#define OFF_SUM   (OFF_SHIFT + 512)
#define OFF_SSQ   (OFF_SUM + 512)
#define SMEM_BYTES (OFF_SSQ + 512)           // 143360

struct alignas(32) SmemTag { float4 a, b; };

// ---------------------------------------------------------------- helpers
__device__ __forceinline__ void async_copy_b128(unsigned int lds_addr, const void* gptr) {
    // GLOBAL_LOAD_ASYNC_TO_LDS_B128: LDS[VDST] <- MEM[VADDR], tracked by ASYNCcnt
    asm volatile("global_load_async_to_lds_b128 %0, %1, off"
                 :: "v"(lds_addr), "v"((unsigned long long)(uintptr_t)gptr)
                 : "memory");
}
__device__ __forceinline__ void wait_async() {
    asm volatile("s_wait_asynccnt 0" ::: "memory");
}
__device__ __forceinline__ v16bf ld_bfrag(const char* rowp, int byteoff) {
    union { v16bf v; float4 f[2]; } u;               // 2x ds_load_b128
    u.f[0] = *(const float4*)(rowp + byteoff);
    u.f[1] = *(const float4*)(rowp + byteoff + 16);
    return u.v;
}
__device__ __forceinline__ unsigned short bfbits(__bf16 h) {
    return __builtin_bit_cast(unsigned short, h);
}

// ---------------------------------------------------------------- small utils
__global__ __launch_bounds__(256) void k_zero_stats(float* __restrict__ stats) {
    int t = threadIdx.x;
    stats[t]       = 0.0f;
    stats[t + 256] = 0.0f;
}

__global__ __launch_bounds__(256) void k_copy(const float4* __restrict__ src,
                                              float4* __restrict__ dst, int n4) {
    int i = blockIdx.x * 256 + threadIdx.x;
    if (i < n4) dst[i] = src[i];
}

// h[dst] += x[src]; one wave per edge, 4 columns per lane
__global__ __launch_bounds__(256) void k_scatter(const float* __restrict__ x,
                                                 const int* __restrict__ ei,
                                                 float* __restrict__ h, int E) {
    int t = blockIdx.x * 256 + threadIdx.x;
    int e = t >> 5;
    if (e >= E) return;
    int lane = t & 31;
    int src = ei[e];
    int dst = ei[E + e];
    const float4 v = ((const float4*)(x + (size_t)src * HD))[lane];
    float* hp = h + (size_t)dst * HD + lane * 4;
    atomicAdd(hp + 0, v.x);
    atomicAdd(hp + 1, v.y);
    atomicAdd(hp + 2, v.z);
    atomicAdd(hp + 3, v.w);
}

// ------------------------------------------------------- fused BN + GEMM + stats
// out = act(in) @ W^T + bias ; act = relu(bn(in)) when in_stats != null.
// fp32 emulated via bf16 split: hi*hi + lo*hi + hi*lo. One block = 128x128 tile.
__global__ __launch_bounds__(256) void k_gemm_bn(
    const float* __restrict__ in,        // [n, 128]
    const float* __restrict__ W,         // [128, 128], W[o][k]
    const float* __restrict__ bias,      // [128]
    const float* __restrict__ in_stats,  // [256] col sum+ssq of in, or null
    const float* __restrict__ in_g,
    const float* __restrict__ in_b,
    float* __restrict__ out,             // [n, 128]
    float* __restrict__ out_stats,       // [256] accumulated, or null
    int n)
{
    extern __shared__ SmemTag smem_raw[];
    char* smem = (char*)smem_raw;

    const int  tid     = threadIdx.x;
    const int  rowbase = blockIdx.x * 128;
    const bool doAct   = (in_stats != nullptr);
    const float inv_n  = 1.0f / (float)n;

    // ---- stage full 128x128 weight matrix as bf16 hi/lo (packed b32 stores)
    for (int idx = tid; idx < 128 * 64; idx += 256) {
        int o = idx >> 6;
        int k = (idx & 63) * 2;
        float2 w = *(const float2*)(W + (size_t)o * HD + k);
        __bf16 h0 = (__bf16)w.x;
        __bf16 h1 = (__bf16)w.y;
        __bf16 l0 = (__bf16)(w.x - (float)h0);
        __bf16 l1 = (__bf16)(w.y - (float)h1);
        unsigned ph = (unsigned)bfbits(h0) | ((unsigned)bfbits(h1) << 16);
        unsigned pl = (unsigned)bfbits(l0) | ((unsigned)bfbits(l1) << 16);
        *(unsigned*)(smem + OFF_WHI + o * WROWB + k * 2) = ph;
        *(unsigned*)(smem + OFF_WLO + o * WROWB + k * 2) = pl;
    }
    // ---- per-K activation affine (bn of previous stage folded into A load)
    if (tid < HD) {
        float sc = 1.0f, sh = 0.0f;
        if (doAct) {
            float mu  = in_stats[tid] * inv_n;
            float var = in_stats[HD + tid] * inv_n - mu * mu;
            sc = in_g[tid] * rsqrtf(var + BN_EPS);
            sh = in_b[tid] - mu * sc;
        }
        *(float*)(smem + OFF_SCALE + tid * 4) = sc;
        *(float*)(smem + OFF_SHIFT + tid * 4) = sh;
        *(float*)(smem + OFF_SUM + tid * 4) = 0.0f;
        *(float*)(smem + OFF_SSQ + tid * 4) = 0.0f;
    }

    // ---- stage 128x128 A tile into LDS via coalesced async loads (ASYNCcnt)
    {
        const int vrows = (n - rowbase < 128) ? (n - rowbase) : 128;
        const unsigned lbase = (unsigned)(uintptr_t)smem;
        for (int i = tid; i < 128 * 32; i += 256) {     // 32 x 16B segs per row
            int r = i >> 5, s = i & 31;
            unsigned loff = (unsigned)(OFF_A + r * AROWB + s * 16);
            if (r < vrows) {
                async_copy_b128(lbase + loff, in + (size_t)(rowbase + r) * HD + s * 4);
            } else {
                *(float4*)(smem + loff) = make_float4(0.f, 0.f, 0.f, 0.f);
            }
        }
        wait_async();
    }
    __syncthreads();

    const int wave = tid >> 5;
    const int lane = tid & 31;
    const int lrow = lane & 15;
    const int hlf  = lane >> 4;

    v8f acc[8] = {};  // eight 16x16 output tiles along N

    const char* aRow    = smem + OFF_A + (size_t)(wave * 16 + lrow) * AROWB;
    const char* whiBase = smem + OFF_WHI;
    const char* wloBase = smem + OFF_WLO;
    const float* sScale = (const float*)(smem + OFF_SCALE);
    const float* sShift = (const float*)(smem + OFF_SHIFT);

    #pragma unroll
    for (int kt = 0; kt < 4; ++kt) {
        // A fragment (contiguous K order, same as the B fragments):
        //  elems 0..7  -> K = kt*32 + 16*hlf + e
        //  elems 8..15 -> K = kt*32 + 16*hlf + 8 + (e-8)
        const int k0 = kt * 32 + hlf * 16;
        const int k1 = kt * 32 + hlf * 16 + 8;
        float buf[16];
        {
            float4 f0 = *(const float4*)(aRow + k0 * 4);
            float4 f1 = *(const float4*)(aRow + k0 * 4 + 16);
            float4 f2 = *(const float4*)(aRow + k1 * 4);
            float4 f3 = *(const float4*)(aRow + k1 * 4 + 16);
            buf[0]=f0.x;  buf[1]=f0.y;  buf[2]=f0.z;  buf[3]=f0.w;
            buf[4]=f1.x;  buf[5]=f1.y;  buf[6]=f1.z;  buf[7]=f1.w;
            buf[8]=f2.x;  buf[9]=f2.y;  buf[10]=f2.z; buf[11]=f2.w;
            buf[12]=f3.x; buf[13]=f3.y; buf[14]=f3.z; buf[15]=f3.w;
        }
        v16bf ahi, alo;
        #pragma unroll
        for (int e = 0; e < 16; ++e) {
            int k = (e < 8) ? (k0 + e) : (k1 + e - 8);
            float v = fmaf(buf[e], sScale[k], sShift[k]);
            if (doAct) v = fmaxf(v, 0.0f);
            __bf16 vh = (__bf16)v;
            ahi[e] = vh;
            alo[e] = (__bf16)(v - (float)vh);
        }
        // B fragments (16-bit 32x16): lane col = lrow, K = kt*32 + 16*hlf + e
        const int ko2 = (kt * 32 + hlf * 16) * 2;
        v16bf bh = ld_bfrag(whiBase + (size_t)lrow * WROWB, ko2);
        v16bf bl = ld_bfrag(wloBase + (size_t)lrow * WROWB, ko2);
        #pragma unroll
        for (int nt = 0; nt < 8; ++nt) {
            v16bf cbh = bh, cbl = bl;    // consume current, prefetch next tile
            if (nt < 7) {
                bh = ld_bfrag(whiBase + (size_t)((nt + 1) * 16 + lrow) * WROWB, ko2);
                bl = ld_bfrag(wloBase + (size_t)((nt + 1) * 16 + lrow) * WROWB, ko2);
            }
            acc[nt] = __builtin_amdgcn_wmma_f32_16x16x32_bf16(
                          false, ahi, false, cbh, (short)0, acc[nt], false, false);
            acc[nt] = __builtin_amdgcn_wmma_f32_16x16x32_bf16(
                          false, alo, false, cbh, (short)0, acc[nt], false, false);
            acc[nt] = __builtin_amdgcn_wmma_f32_16x16x32_bf16(
                          false, ahi, false, cbl, (short)0, acc[nt], false, false);
        }
    }

    // ---- epilogue: D layout -> lane row = wave*16 + 8*hlf + r, col = nt*16 + lrow
    const int rbase = rowbase + wave * 16 + hlf * 8;
    float* sSum = (float*)(smem + OFF_SUM);
    float* sSsq = (float*)(smem + OFF_SSQ);
    #pragma unroll
    for (int nt = 0; nt < 8; ++nt) {
        const int col = nt * 16 + lrow;
        const float b = bias[col];
        float psum = 0.0f, pssq = 0.0f;
        #pragma unroll
        for (int r = 0; r < 8; ++r) {
            int orow = rbase + r;
            if (orow < n) {
                float v = acc[nt][r] + b;
                out[(size_t)orow * HD + col] = v;
                psum += v;
                pssq += v * v;
            }
        }
        if (out_stats) {
            atomicAdd(&sSum[col], psum);
            atomicAdd(&sSsq[col], pssq);
        }
    }
    if (out_stats) {
        __syncthreads();
        if (tid < HD) {
            atomicAdd(&out_stats[tid],      sSum[tid]);
            atomicAdd(&out_stats[HD + tid], sSsq[tid]);
        }
    }
}

// -------------------------------- bn2 + bn3 + relu (bn3 folded analytically)
// bn2-out has column mean b2 and variance s2^2*var2 exactly -> no extra pass.
// Also writes hinit (next layer's h base) to skip a separate copy kernel.
__global__ __launch_bounds__(256) void k_bn_final(
    const float* __restrict__ y,       // fc2 out + bias, [n,128]
    const float* __restrict__ stats,   // [256] col sum+ssq of y
    const float* __restrict__ g2, const float* __restrict__ b2,
    const float* __restrict__ g3, const float* __restrict__ b3,
    float* __restrict__ xout, float* __restrict__ hinit, int n)
{
    int idx = blockIdx.x * 256 + threadIdx.x;
    if (idx >= n * HD) return;
    int col = idx & (HD - 1);
    float inv_n = 1.0f / (float)n;
    float mu  = stats[col] * inv_n;
    float var = stats[HD + col] * inv_n - mu * mu;
    float s2  = g2[col] * rsqrtf(var + BN_EPS);
    float sh2 = b2[col] - mu * s2;
    float s3  = g3[col] * rsqrtf(s2 * s2 * var + BN_EPS);
    float S   = s2 * s3;
    float T   = (sh2 - b2[col]) * s3 + b3[col];
    float v   = fmaxf(fmaf(y[idx], S, T), 0.0f);
    xout[idx]  = v;
    hinit[idx] = v;
}

// --------------------------------------------------------------------- launcher
extern "C" void kernel_launch(void* const* d_in, const int* in_sizes, int n_in,
                              void* d_out, int out_size, void* d_ws, size_t ws_size,
                              hipStream_t stream) {
    const float* x     = (const float*)d_in[0];
    const int*   ei    = (const int*)d_in[1];
    const float* fc1_w = (const float*)d_in[2];
    const float* fc1_b = (const float*)d_in[3];
    const float* bn1_g = (const float*)d_in[4];
    const float* bn1_b = (const float*)d_in[5];
    const float* fc2_w = (const float*)d_in[6];
    const float* fc2_b = (const float*)d_in[7];
    const float* bn2_g = (const float*)d_in[8];
    const float* bn2_b = (const float*)d_in[9];
    const float* bn3_g = (const float*)d_in[10];
    const float* bn3_b = (const float*)d_in[11];
    const float* fc_w  = (const float*)d_in[12];
    const float* fc_b  = (const float*)d_in[13];

    const int N = in_sizes[0] / HD;
    const int E = in_sizes[1] / 2;
    const int L = in_sizes[2] / (HD * HD);

    float* ws   = (float*)d_ws;
    float* hbuf = ws;                        // [N,128]; reused as y2
    float* y1   = ws + (size_t)N * HD;       // [N,128]
    float* xbuf = ws + 2 * (size_t)N * HD;   // [N,128]
    float* st   = ws + 3 * (size_t)N * HD;   // [512]: stats1(256) + stats2(256)
    float* st1  = st;
    float* st2  = st + 256;

    (void)hipFuncSetAttribute((const void*)k_gemm_bn,
                              hipFuncAttributeMaxDynamicSharedMemorySize, SMEM_BYTES);

    dim3 gGemm((N + 127) / 128);
    const int n4       = (N * HD) / 4;
    const int cpBlocks = (n4 + 255) / 256;
    const int scBlocks = (int)(((long long)E * 32 + 255) / 256);
    const int ewBlocks = (N * HD + 255) / 256;

    const float* cur = x;
    for (int i = 0; i < L; ++i) {
        k_zero_stats<<<1, 256, 0, stream>>>(st);
        if (i == 0)  // later layers: h base pre-written by k_bn_final
            k_copy<<<cpBlocks, 256, 0, stream>>>((const float4*)cur, (float4*)hbuf, n4);
        k_scatter<<<scBlocks, 256, 0, stream>>>(cur, ei, hbuf, E);
        // y1 = h @ fc1^T + b1 ; accumulate stats1
        k_gemm_bn<<<gGemm, 256, SMEM_BYTES, stream>>>(
            hbuf, fc1_w + (size_t)i * HD * HD, fc1_b + i * HD,
            nullptr, nullptr, nullptr, y1, st1, N);
        // y2 = relu(bn1(y1)) @ fc2^T + b2 ; accumulate stats2 (y2 overwrites hbuf)
        k_gemm_bn<<<gGemm, 256, SMEM_BYTES, stream>>>(
            y1, fc2_w + (size_t)i * HD * HD, fc2_b + i * HD,
            st1, bn1_g + i * HD, bn1_b + i * HD, hbuf, st2, N);
        // x = relu(bn3(bn2(y2))); also seeds next layer's h base
        k_bn_final<<<ewBlocks, 256, 0, stream>>>(
            hbuf, st2, bn2_g + i * HD, bn2_b + i * HD,
            bn3_g + i * HD, bn3_b + i * HD, xbuf, hbuf, N);
        cur = xbuf;
    }
    // out = x @ fc_w^T + fc_b
    k_gemm_bn<<<gGemm, 256, SMEM_BYTES, stream>>>(
        cur, fc_w, fc_b, nullptr, nullptr, nullptr, (float*)d_out, nullptr, N);
}
